// Baseline_1709396984290
// MI455X (gfx1250) — matrix operand order repaired and checked by reading, not compiled, not run on hardware
//
#include <hip/hip_runtime.h>
#include <hip/hip_bf16.h>
#include <cstdint>

// ---------------------------------------------------------------------------
// Model dims (fixed by the reference)
// ---------------------------------------------------------------------------
#define BN_   32          // batch
#define SN_   64          // source length
#define TN_   64          // target length
#define EN_   512         // embedding dim
#define HN_   512         // hidden dim
#define G3_   1536        // 3*H (GRU gate width)
#define VT_   32000       // target vocab

typedef __attribute__((ext_vector_type(16))) __bf16 bf16x16;
typedef __attribute__((ext_vector_type(8)))  float  f32x8;
typedef unsigned int u32;
typedef u32 u32x4 __attribute__((ext_vector_type(4)));
typedef u32 u32x8 __attribute__((ext_vector_type(8)));

enum { ACT_NONE = 0, ACT_RELU = 1, ACT_TANH = 2 };

// ---------------------------------------------------------------------------
// Tensor Data Mover: stage a 32-row x 256-col bf16 tile (row stride K elems)
// from global memory into LDS. TDM pad feature inserts 4 DWORDs after each
// 128-DWORD (256 bf16) row -> padded LDS row stride 264 bf16 = 528B
// (16B aligned per row, conflict-free ds_load_b128 fragment reads).
// D# layout per CDNA5 ISA ch.8 (group0: 4 SGPRs, group1: 8 SGPRs).
// ---------------------------------------------------------------------------
#define TDM_CHUNK 256     // K elements staged per DMA
#define TDM_ROWE  264     // padded LDS row stride in bf16 elements

__device__ __forceinline__ void tdm_load_tile(u32 lds_addr, const __bf16* gptr, int K)
{
    const unsigned long long ga = (unsigned long long)(uintptr_t)gptr;
    u32x4 g0;
    g0[0] = 1u;                                   // count=1, is_restore=0
    g0[1] = lds_addr;                             // LDS byte address of tile
    g0[2] = (u32)ga;                              // global_addr[31:0]
    g0[3] = (u32)(ga >> 32) | (2u << 30);         // global_addr[56:32] | type=2
    u32x8 g1;
    g1[0] = (1u << 16)                            // data_size = 2 bytes
          | (1u << 20)                            // pad_enable
          | (6u << 22)                            // pad_interval = 128 DWORDs
          | (3u << 25);                           // pad_amount  = 4 DWORDs
    g1[1] = (u32)TDM_CHUNK << 16;                 // tensor_dim0 = 256
    g1[2] = 32u << 16;                            // tensor_dim1 = 32
    g1[3] = (u32)TDM_CHUNK << 16;                 // tile_dim0 = 256
    g1[4] = 32u;                                  // tile_dim1 = 32, tile_dim2 = 0
    g1[5] = (u32)K;                               // tensor_dim0_stride[31:0] = K
    g1[6] = 0u;                                   // stride hi / dim1_stride lo
    g1[7] = 0u;
    asm volatile("tensor_load_to_lds %0, %1" :: "s"(g0), "s"(g1) : "memory");
}

// ---------------------------------------------------------------------------
// WMMA bf16 GEMM:  C[M,N](f32) = act( A[M,K](bf16) * Bw[N,K]^T(bf16) + bias )
// Requirements: M % 32 == 0, N % 128 == 0, K % 256 == 0.
// Block = 256 threads (8 waves). Each wave computes a 32x16 output strip
// (two 16x16 C tiles sharing one B fragment). The 32xK A block (shared by
// all 8 waves) is staged into LDS by the Tensor Data Mover, double-buffered
// in 32x256 chunks; the B weight stream keeps the global-memory path.
// Fragment layouts follow the CDNA5 ISA 16-bit A(16x32)/B(32x16)/C striping.
// ---------------------------------------------------------------------------
union FragAB { bf16x16 v; uint4 u[2]; };

__global__ __launch_bounds__(256)
void k_gemm_bf16(const __bf16* __restrict__ A,   // [M,K] row-major
                 const __bf16* __restrict__ Bw,  // [N,K] row-major (weights)
                 const float*  __restrict__ bias,// [N] or nullptr
                 float*        __restrict__ C,   // row stride = ldc
                 __bf16*       __restrict__ Cbf, // optional bf16 mirror (ldc)
                 int M, int N, int K, int ldc, int act)
{
    __shared__ __align__(16) __bf16 smA[2][32 * TDM_ROWE];  // 2 x 16.5KB

    const int lane  = threadIdx.x & 31;
    const int wave  = threadIdx.x >> 5;
    const int m0    = blockIdx.y * 32;                 // block M base
    const int ncol0 = blockIdx.x * 128 + wave * 16;    // this wave's N-tile

    const int mrow = lane & 15;    // A: M row in tile ; B: N col in tile
    const int kh   = lane >> 4;    // lane half selects K group

    f32x8 acc0 = {};  // M rows m0 .. m0+15
    f32x8 acc1 = {};  // M rows m0+16 .. m0+31

    const __bf16* __restrict__ Brow = Bw + (size_t)(ncol0 + mrow) * K;
    const __bf16* __restrict__ Ablk = A + (size_t)m0 * K;

    const int nch = K / TDM_CHUNK;
    if (wave == 0)
        tdm_load_tile((u32)(uintptr_t)&smA[0][0], Ablk, K);   // chunk 0

    for (int c = 0; c < nch; ++c) {
        if (wave == 0) {
            if (c + 1 < nch) {
                tdm_load_tile((u32)(uintptr_t)&smA[(c + 1) & 1][0],
                              Ablk + (size_t)(c + 1) * TDM_CHUNK, K);
                __builtin_amdgcn_s_wait_tensorcnt(1);  // chunk c done (in-order)
            } else {
                __builtin_amdgcn_s_wait_tensorcnt(0);
            }
        }
        __syncthreads();   // chunk c visible to all waves

        const __bf16* sA0 = &smA[c & 1][(size_t)mrow * TDM_ROWE];
        const __bf16* sA1 = sA0 + 16 * TDM_ROWE;
        const __bf16* bp0 = Brow + c * TDM_CHUNK + 8 * kh;

        for (int k0 = 0; k0 < TDM_CHUNK; k0 += 32) {
            FragAB a0, a1, b;
            // A 16x32 fragment from LDS: lanes<16 hold K {0..7,16..23},
            // lanes>=16 hold K {8..15,24..31}
            a0.u[0] = *(const uint4*)(sA0 + k0 +      8 * kh);
            a0.u[1] = *(const uint4*)(sA0 + k0 + 16 + 8 * kh);
            a1.u[0] = *(const uint4*)(sA1 + k0 +      8 * kh);
            a1.u[1] = *(const uint4*)(sA1 + k0 + 16 + 8 * kh);
            // B 32x16 fragment: lanes<16 hold K {0..7,16..23}, lanes>=16 K {8..15,24..31}
            const __bf16* bp = bp0 + k0;
            b.u[0] = *(const uint4*)(bp);
            b.u[1] = *(const uint4*)(bp + 16);

            if (c * TDM_CHUNK + k0 + 32 < K)  // prefetch next weight K-chunk
                __builtin_prefetch(Brow + c * TDM_CHUNK + k0 + 32, 0, 1);

            acc0 = __builtin_amdgcn_wmma_f32_16x16x32_bf16(
                       false, a0.v, false, b.v, (short)0, acc0, false, false);
            acc1 = __builtin_amdgcn_wmma_f32_16x16x32_bf16(
                       false, a1.v, false, b.v, (short)0, acc1, false, false);
        }
        __syncthreads();   // chunk consumed before TDM overwrites buffer
    }

    const int   col = ncol0 + (lane & 15);
    const float bv  = bias ? bias[col] : 0.0f;

#pragma unroll
    for (int j = 0; j < 8; ++j) {
        const int mloc = j + 8 * kh;   // C tile row for VGPR j
        float v0 = acc0[j] + bv;
        float v1 = acc1[j] + bv;
        if (act == ACT_RELU)      { v0 = fmaxf(v0, 0.f); v1 = fmaxf(v1, 0.f); }
        else if (act == ACT_TANH) { v0 = tanhf(v0);      v1 = tanhf(v1); }
        C[(size_t)(m0 + mloc)      * ldc + col] = v0;
        C[(size_t)(m0 + 16 + mloc) * ldc + col] = v1;
        if (Cbf) {
            Cbf[(size_t)(m0 + mloc)      * ldc + col] = (__bf16)v0;
            Cbf[(size_t)(m0 + 16 + mloc) * ldc + col] = (__bf16)v1;
        }
    }
}

// ---------------------------------------------------------------------------
// Fused GRU gate:  r = sig(gi_r+gh_r); z = sig(gi_z+gh_z);
//                  n = tanh(gi_n + r*gh_n); h = (1-z)*n + z*h_prev
// gi/gh already include their biases (added in the GEMM).
// ---------------------------------------------------------------------------
__device__ __forceinline__ float sigmoid_(float x) { return 1.0f / (1.0f + expf(-x)); }

__global__ __launch_bounds__(256)
void k_gru_gate(const float* __restrict__ gi, const float* __restrict__ gh,
                const float* __restrict__ hprev,
                float* __restrict__ hout, __bf16* __restrict__ houtbf,
                float* __restrict__ yout, __bf16* __restrict__ youtbf,
                int ystride, int yoff,
                float* __restrict__ accum, int accstride, int accoff,
                __bf16* __restrict__ cat2, const __bf16* __restrict__ et)
{
    const int idx = blockIdx.x * blockDim.x + threadIdx.x;
    if (idx >= BN_ * HN_) return;
    const int b = idx >> 9;         // / 512
    const int j = idx & 511;        // % 512

    const float* gib = gi + (size_t)b * G3_;
    const float* ghb = gh + (size_t)b * G3_;

    const float r = sigmoid_(gib[j] + ghb[j]);
    const float z = sigmoid_(gib[HN_ + j] + ghb[HN_ + j]);
    const float n = tanhf(gib[2 * HN_ + j] + r * ghb[2 * HN_ + j]);
    const float h = (1.0f - z) * n + z * hprev[idx];

    hout[idx]   = h;
    houtbf[idx] = (__bf16)h;
    if (yout)   yout  [(size_t)b * ystride + yoff + j] = h;
    if (youtbf) youtbf[(size_t)b * ystride + yoff + j] = (__bf16)h;
    if (accum)  accum [(size_t)b * accstride + accoff + j] += h;
    if (cat2) {
        cat2[(size_t)b * 1024 + j]        = (__bf16)h;
        cat2[(size_t)b * 1024 + 512 + j]  = et[(size_t)b * 512 + j];
    }
}

// ---------------------------------------------------------------------------
// Fused attention step (one block per batch row, 256 thr = 8 waves):
//   scores[s] = we . tanh(pq[b] + proj_keys[s,b]);  masked softmax over s;
//   context c = sum_s alpha[s]*enc_out[s,b];  xcat = [bf16(c), e_t]
// ---------------------------------------------------------------------------
__global__ __launch_bounds__(256)
void k_attention(const float* __restrict__ pq,        // [B,512]
                 const float* __restrict__ proj_keys, // [S,B,512]
                 const float* __restrict__ we,        // [512]
                 const unsigned char* __restrict__ mask, // [B,S] bool
                 const float* __restrict__ enc_out,   // [S,B,1024]
                 const __bf16* __restrict__ es_t,     // [B,512]
                 __bf16* __restrict__ xcat)           // [B,1536]
{
    const int b    = blockIdx.x;
    const int lane = threadIdx.x & 31;
    const int wave = threadIdx.x >> 5;

    __shared__ float sc[SN_];
    __shared__ float alpha[SN_];

    const float* pqb = pq + (size_t)b * HN_;
    for (int s = wave; s < SN_; s += 8) {
        const float* pk = proj_keys + ((size_t)s * BN_ + b) * HN_;
        float acc = 0.0f;
        for (int h = lane; h < HN_; h += 32)
            acc += we[h] * tanhf(pqb[h] + pk[h]);
        for (int off = 16; off; off >>= 1) acc += __shfl_xor(acc, off, 32);
        if (lane == 0) sc[s] = mask[b * SN_ + s] ? acc : -1.0e9f;
    }
    __syncthreads();

    if (wave == 0) {   // wave32 softmax over 64 scores (2 per lane)
        float v0 = sc[lane], v1 = sc[lane + 32];
        float m = fmaxf(v0, v1);
        for (int off = 16; off; off >>= 1) m = fmaxf(m, __shfl_xor(m, off, 32));
        float e0 = expf(v0 - m), e1 = expf(v1 - m);
        float sum = e0 + e1;
        for (int off = 16; off; off >>= 1) sum += __shfl_xor(sum, off, 32);
        const float inv = 1.0f / sum;
        alpha[lane]      = e0 * inv;
        alpha[lane + 32] = e1 * inv;
    }
    __syncthreads();

    for (int d = threadIdx.x; d < 2 * HN_; d += 256) {
        float c = 0.0f;
        for (int s = 0; s < SN_; ++s)
            c += alpha[s] * enc_out[((size_t)s * BN_ + b) * (2 * HN_) + d];
        xcat[(size_t)b * G3_ + d] = (__bf16)c;
    }
    for (int j = threadIdx.x; j < HN_; j += 256)
        xcat[(size_t)b * G3_ + 2 * HN_ + j] = es_t[(size_t)b * HN_ + j];
}

// ---------------------------------------------------------------------------
// Small utility kernels
// ---------------------------------------------------------------------------
__global__ void k_f32_to_bf16(const float* __restrict__ s, __bf16* __restrict__ d, size_t n)
{
    for (size_t i = blockIdx.x * (size_t)blockDim.x + threadIdx.x; i < n;
         i += (size_t)gridDim.x * blockDim.x)
        d[i] = (__bf16)s[i];
}

// out[(t*B + b)*E + e] = bf16(emb[ids[b*S + t]*E + e])   (time-major gather)
__global__ void k_embed_gather(const float* __restrict__ emb, const int* __restrict__ ids,
                               __bf16* __restrict__ out, int Bn, int Sn, int En)
{
    const size_t total = (size_t)Bn * Sn * En;
    for (size_t i = blockIdx.x * (size_t)blockDim.x + threadIdx.x; i < total;
         i += (size_t)gridDim.x * blockDim.x) {
        const int e = (int)(i % En);
        const size_t r = i / En;
        const int b = (int)(r / Sn);
        const int t = (int)(r % Sn);
        out[((size_t)t * Bn + b) * En + e] =
            (__bf16)emb[(size_t)ids[(size_t)b * Sn + t] * En + e];
    }
}

__global__ void k_zero_state(float* __restrict__ h, __bf16* __restrict__ hb, int n)
{
    const int i = blockIdx.x * blockDim.x + threadIdx.x;
    if (i < n) { h[i] = 0.0f; if (hb) hb[i] = (__bf16)0.0f; }
}

__global__ void k_copy_state(const float* __restrict__ s, const __bf16* __restrict__ sb,
                             float* __restrict__ d, __bf16* __restrict__ db, int n)
{
    const int i = blockIdx.x * blockDim.x + threadIdx.x;
    if (i < n) { d[i] = s[i]; db[i] = sb[i]; }
}

__global__ void k_scale_to_bf16(const float* __restrict__ s, __bf16* __restrict__ d,
                                int n, float scale)
{
    const int i = blockIdx.x * blockDim.x + threadIdx.x;
    if (i < n) d[i] = (__bf16)(s[i] * scale);
}

// ---------------------------------------------------------------------------
// Host orchestration
// ---------------------------------------------------------------------------
extern "C" void kernel_launch(void* const* d_in, const int* in_sizes, int n_in,
                              void* d_out, int out_size, void* d_ws, size_t ws_size,
                              hipStream_t stream)
{
    enum { P_EMB_X = 0, P_EMB_Y, P_INF_WI_F, P_INF_WH_F, P_INF_BI_F, P_INF_BH_F,
           P_INF_WI_B, P_INF_WH_B, P_INF_BI_B, P_INF_BH_B,
           P_U_HID_W, P_U_HID_B, P_U_OUT_W, P_U_OUT_B,
           P_ENC_INIT_W, P_ENC_INIT_B,
           P_ENC_WI_F, P_ENC_WH_F, P_ENC_BI_F, P_ENC_BH_F,
           P_ENC_WI_B, P_ENC_WH_B, P_ENC_BI_B, P_ENC_BH_B,
           P_ATT_WK, P_ATT_WQ, P_ATT_WE,
           P_DEC_INIT_W, P_DEC_INIT_B,
           P_DEC_WI, P_DEC_WH, P_DEC_BI, P_DEC_BH,
           P_OUT_W, P_OUT_B, P_X, P_XMASK, P_PREV };

    auto F = [&](int i) { return (const float*)d_in[i]; };
    auto I = [&](int i) { return (const int*)d_in[i]; };

    // bump allocator over workspace (256B aligned)
    size_t off = 0;
    auto alloc = [&](size_t bytes) -> void* {
        off = (off + 255) & ~(size_t)255;
        void* p = (char*)d_ws + off;
        off += bytes;
        return p;
    };

    const int B = BN_, S = SN_, T = TN_, E = EN_, H = HN_, G3 = G3_, VT = VT_;

    // ---- bf16 weight copies -------------------------------------------------
    auto cvt = [&](int idx, size_t n) -> __bf16* {
        __bf16* p = (__bf16*)alloc(n * 2);
        size_t blocks = (n + 255) / 256; if (blocks > 4096) blocks = 4096;
        k_f32_to_bf16<<<dim3((unsigned)blocks), dim3(256), 0, stream>>>(F(idx), p, n);
        return p;
    };
    __bf16* wInfWiF = cvt(P_INF_WI_F, (size_t)G3 * E);
    __bf16* wInfWhF = cvt(P_INF_WH_F, (size_t)G3 * H);
    __bf16* wInfWiB = cvt(P_INF_WI_B, (size_t)G3 * E);
    __bf16* wInfWhB = cvt(P_INF_WH_B, (size_t)G3 * H);
    __bf16* wUhid   = cvt(P_U_HID_W,  (size_t)H * 2 * H);
    __bf16* wUout   = cvt(P_U_OUT_W,  (size_t)H * H);
    __bf16* wEncIni = cvt(P_ENC_INIT_W, (size_t)H * H);
    __bf16* wEncWiF = cvt(P_ENC_WI_F, (size_t)G3 * E);
    __bf16* wEncWhF = cvt(P_ENC_WH_F, (size_t)G3 * H);
    __bf16* wEncWiB = cvt(P_ENC_WI_B, (size_t)G3 * E);
    __bf16* wEncWhB = cvt(P_ENC_WH_B, (size_t)G3 * H);
    __bf16* wAttWk  = cvt(P_ATT_WK,   (size_t)H * 2 * H);
    __bf16* wAttWq  = cvt(P_ATT_WQ,   (size_t)H * H);
    __bf16* wDecIni = cvt(P_DEC_INIT_W, (size_t)H * H);
    __bf16* wDecWi  = cvt(P_DEC_WI,   (size_t)G3 * (E + 2 * H));
    __bf16* wDecWh  = cvt(P_DEC_WH,   (size_t)G3 * H);
    __bf16* wOutW   = cvt(P_OUT_W,    (size_t)VT * (H + E));

    // ---- activation buffers -------------------------------------------------
    __bf16* xs_bf   = (__bf16*)alloc((size_t)S * B * E * 2);   // [S,B,E]
    __bf16* es_bf   = (__bf16*)alloc((size_t)T * B * E * 2);   // [T,B,E]
    float*  giAll   = (float*) alloc((size_t)S * B * G3 * 4);  // [S,B,3H]
    float*  gh      = (float*) alloc((size_t)B * G3 * 4);
    float*  gi_dec  = (float*) alloc((size_t)B * G3 * 4);
    float*  h_f32   = (float*) alloc((size_t)B * H * 4);
    __bf16* h_bf    = (__bf16*)alloc((size_t)B * H * 2);
    float*  hx_sum  = (float*) alloc((size_t)B * 2 * H * 4);
    __bf16* hx_bf   = (__bf16*)alloc((size_t)B * 2 * H * 2);
    float*  t1_f    = (float*) alloc((size_t)B * H * 4);
    __bf16* t1_bf   = (__bf16*)alloc((size_t)B * H * 2);
    float*  z_f     = (float*) alloc((size_t)B * H * 4);
    __bf16* z_bf    = (__bf16*)alloc((size_t)B * H * 2);
    float*  he_f    = (float*) alloc((size_t)B * H * 4);
    __bf16* he_bf   = (__bf16*)alloc((size_t)B * H * 2);
    float*  hd_f    = (float*) alloc((size_t)B * H * 4);
    __bf16* hd_bf   = (__bf16*)alloc((size_t)B * H * 2);
    float*  encOutF = (float*) alloc((size_t)S * B * 2 * H * 4); // [S,B,2H]
    __bf16* encOutB = (__bf16*)alloc((size_t)S * B * 2 * H * 2);
    float*  projK   = (float*) alloc((size_t)S * B * H * 4);     // [S,B,H]
    float*  pq      = (float*) alloc((size_t)B * H * 4);
    __bf16* xcat_bf = (__bf16*)alloc((size_t)B * G3 * 2);
    __bf16* cat2_bf = (__bf16*)alloc((size_t)B * (H + E) * 2);
    (void)ws_size; (void)n_in; (void)in_sizes; (void)out_size;

    auto gemm = [&](const __bf16* A, const __bf16* Bw, const float* bias,
                    float* C, __bf16* Cbf, int M, int N, int K, int ldc, int act) {
        k_gemm_bf16<<<dim3(N / 128, M / 32), dim3(256), 0, stream>>>(
            A, Bw, bias, C, Cbf, M, N, K, ldc, act);
    };

    // ---- embeddings (time-major, bf16) -------------------------------------
    k_embed_gather<<<dim3(4096), dim3(256), 0, stream>>>(F(P_EMB_X), I(P_X), xs_bf, B, S, E);
    k_embed_gather<<<dim3(4096), dim3(256), 0, stream>>>(F(P_EMB_Y), I(P_PREV), es_bf, B, T, E);

    // ---- inference-net biGRU, accumulating sum over time into hx_sum -------
    k_zero_state<<<dim3((B * 2 * H + 255) / 256), dim3(256), 0, stream>>>(hx_sum, nullptr, B * 2 * H);

    // forward direction
    gemm(xs_bf, wInfWiF, F(P_INF_BI_F), giAll, nullptr, S * B, G3, E, G3, ACT_NONE);
    k_zero_state<<<dim3((B * H + 255) / 256), dim3(256), 0, stream>>>(h_f32, h_bf, B * H);
    for (int t = 0; t < S; ++t) {
        gemm(h_bf, wInfWhF, F(P_INF_BH_F), gh, nullptr, B, G3, H, G3, ACT_NONE);
        k_gru_gate<<<dim3(64), dim3(256), 0, stream>>>(
            giAll + (size_t)t * B * G3, gh, h_f32, h_f32, h_bf,
            nullptr, nullptr, 0, 0, hx_sum, 2 * H, 0, nullptr, nullptr);
    }
    // backward direction
    gemm(xs_bf, wInfWiB, F(P_INF_BI_B), giAll, nullptr, S * B, G3, E, G3, ACT_NONE);
    k_zero_state<<<dim3((B * H + 255) / 256), dim3(256), 0, stream>>>(h_f32, h_bf, B * H);
    for (int t = S - 1; t >= 0; --t) {
        gemm(h_bf, wInfWhB, F(P_INF_BH_B), gh, nullptr, B, G3, H, G3, ACT_NONE);
        k_gru_gate<<<dim3(64), dim3(256), 0, stream>>>(
            giAll + (size_t)t * B * G3, gh, h_f32, h_f32, h_bf,
            nullptr, nullptr, 0, 0, hx_sum, 2 * H, H, nullptr, nullptr);
    }
    // mean over time -> bf16
    k_scale_to_bf16<<<dim3((B * 2 * H + 255) / 256), dim3(256), 0, stream>>>(
        hx_sum, hx_bf, B * 2 * H, 1.0f / (float)S);

    // ---- bottleneck MLP + init states --------------------------------------
    gemm(hx_bf, wUhid,   F(P_U_HID_B),    t1_f, t1_bf, B, H, 2 * H, H, ACT_RELU);
    gemm(t1_bf, wUout,   F(P_U_OUT_B),    z_f,  z_bf,  B, H, H,     H, ACT_NONE);
    gemm(z_bf,  wEncIni, F(P_ENC_INIT_B), he_f, he_bf, B, H, H,     H, ACT_TANH);
    gemm(z_bf,  wDecIni, F(P_DEC_INIT_B), hd_f, hd_bf, B, H, H,     H, ACT_TANH);

    // ---- encoder biGRU (h0 = he), writing enc_out [S,B,2H] -----------------
    gemm(xs_bf, wEncWiF, F(P_ENC_BI_F), giAll, nullptr, S * B, G3, E, G3, ACT_NONE);
    k_copy_state<<<dim3((B * H + 255) / 256), dim3(256), 0, stream>>>(he_f, he_bf, h_f32, h_bf, B * H);
    for (int t = 0; t < S; ++t) {
        gemm(h_bf, wEncWhF, F(P_ENC_BH_F), gh, nullptr, B, G3, H, G3, ACT_NONE);
        k_gru_gate<<<dim3(64), dim3(256), 0, stream>>>(
            giAll + (size_t)t * B * G3, gh, h_f32, h_f32, h_bf,
            encOutF + (size_t)t * B * 2 * H, encOutB + (size_t)t * B * 2 * H, 2 * H, 0,
            nullptr, 0, 0, nullptr, nullptr);
    }
    gemm(xs_bf, wEncWiB, F(P_ENC_BI_B), giAll, nullptr, S * B, G3, E, G3, ACT_NONE);
    k_copy_state<<<dim3((B * H + 255) / 256), dim3(256), 0, stream>>>(he_f, he_bf, h_f32, h_bf, B * H);
    for (int t = S - 1; t >= 0; --t) {
        gemm(h_bf, wEncWhB, F(P_ENC_BH_B), gh, nullptr, B, G3, H, G3, ACT_NONE);
        k_gru_gate<<<dim3(64), dim3(256), 0, stream>>>(
            giAll + (size_t)t * B * G3, gh, h_f32, h_f32, h_bf,
            encOutF + (size_t)t * B * 2 * H, encOutB + (size_t)t * B * 2 * H, 2 * H, H,
            nullptr, 0, 0, nullptr, nullptr);
    }

    // ---- attention keys: proj_keys = enc_out @ Wk^T ------------------------
    gemm(encOutB, wAttWk, nullptr, projK, nullptr, S * B, H, 2 * H, H, ACT_NONE);

    // ---- decoder loop -------------------------------------------------------
    k_copy_state<<<dim3((B * H + 255) / 256), dim3(256), 0, stream>>>(hd_f, hd_bf, h_f32, h_bf, B * H);
    float* out = (float*)d_out;   // logits [B,T,VT]
    for (int t = 0; t < T; ++t) {
        gemm(h_bf, wAttWq, nullptr, pq, nullptr, B, H, H, H, ACT_NONE);
        k_attention<<<dim3(B), dim3(256), 0, stream>>>(
            pq, projK, F(P_ATT_WE), (const unsigned char*)d_in[P_XMASK],
            encOutF, es_bf + (size_t)t * B * E, xcat_bf);
        gemm(xcat_bf, wDecWi, F(P_DEC_BI), gi_dec, nullptr, B, G3, E + 2 * H, G3, ACT_NONE);
        gemm(h_bf,    wDecWh, F(P_DEC_BH), gh,     nullptr, B, G3, H,         G3, ACT_NONE);
        k_gru_gate<<<dim3(64), dim3(256), 0, stream>>>(
            gi_dec, gh, h_f32, h_f32, h_bf,
            nullptr, nullptr, 0, 0, nullptr, 0, 0,
            cat2_bf, es_bf + (size_t)t * B * E);
        // logits_t = [h, e_t] @ out_W^T + out_b   -> out[b, t, :]
        gemm(cat2_bf, wOutW, F(P_OUT_B), out + (size_t)t * VT, nullptr,
             B, VT, H + E, T * VT, ACT_NONE);
    }
}
